// SphericalModelDeep_39728447488187
// MI455X (gfx1250) — compile-verified, matrix operand order checked by reading, not run on hardware
//
#include <hip/hip_runtime.h>
#include <hip/hip_bf16.h>

// ---------------------------------------------------------------------------
// CDNA5 / gfx1250 spherical-CNN pipeline.
// All heavy math as bf16 WMMA GEMMs (f32 accumulate), L2-resident chain,
// TDM async tensor loads staging tiles into LDS (double buffered).
// 32x32 wave tiles: 4 v_wmma per k-step per wave (2x the LDS reuse of 16x32).
// GEMM epilogue emits relu'd bf16 activations directly (no convert pass).
// ---------------------------------------------------------------------------

typedef __attribute__((ext_vector_type(16))) __bf16       v16bf;
typedef __attribute__((ext_vector_type(8)))  float        v8f;
typedef __attribute__((ext_vector_type(4)))  unsigned int v4u;
typedef __attribute__((ext_vector_type(8)))  int          v8i;
typedef __attribute__((ext_vector_type(4)))  int          v4i;

#define BM 64
#define BN 128
#define BK 32

#if defined(__has_builtin)
#if __has_builtin(__builtin_amdgcn_tensor_load_to_lds)
#define HAVE_TDM 1
#endif
#if __has_builtin(__builtin_amdgcn_s_wait_tensorcnt)
#define HAVE_TENSORCNT 1
#endif
#endif
#ifndef HAVE_TDM
#define HAVE_TDM 0
#endif
#ifndef HAVE_TENSORCNT
#define HAVE_TENSORCNT 0
#endif

// float -> bf16 (round to nearest even) as raw u16.
__device__ inline unsigned short f2bf(float f) {
    unsigned u = __builtin_bit_cast(unsigned, f);
    unsigned r = u + 0x7FFFu + ((u >> 16) & 1u);
    return (unsigned short)(r >> 16);
}

// ---------------------------------------------------------------------------
// TDM staging: one 2D tile (tileRows x BK bf16 elems) global -> LDS.
// D# packing per CDNA5 ISA (groups 0/1; 2/3 zero = 2D).
// ---------------------------------------------------------------------------
#if HAVE_TDM
__device__ inline void tdm_stage(unsigned short* dstLds,
                                 const unsigned short* gsrc,
                                 int tileRows, int ld) {
    unsigned ldsAddr = (unsigned)(unsigned long long)(void*)dstLds;
    unsigned long long ga = (unsigned long long)gsrc;

    v4u g0;
    g0[0] = 1u;                                                 // count=1
    g0[1] = ldsAddr;                                            // LDS byte addr
    g0[2] = (unsigned)ga;                                       // global[31:0]
    g0[3] = (unsigned)((ga >> 32) & 0x1FFFFFFu) | 0x80000000u;  // [56:32]|type=2

    unsigned td0 = (unsigned)ld;      // tensor_dim0 = row length (elems)
    unsigned td1 = 0x8000u;           // tensor_dim1 (large)
    v8i g1;
    g1[0] = (int)(1u << 16);          // workgroup_mask=0, data_size=1 (2B)
    g1[1] = (int)((td0 & 0xFFFFu) << 16);
    g1[2] = (int)(((td0 >> 16) & 0xFFFFu) | ((td1 & 0xFFFFu) << 16));
    g1[3] = (int)(((td1 >> 16) & 0xFFFFu) | ((unsigned)BK << 16));  // tile_dim0=BK
    g1[4] = (int)((unsigned)tileRows & 0xFFFFu);                    // tile_dim1
    g1[5] = (int)td0;                 // tensor_dim0_stride = ld
    g1[6] = 0;
    g1[7] = 0;
    v4i z4 = {0, 0, 0, 0};
#if __clang_major__ >= 23
    v8i z8 = {0, 0, 0, 0, 0, 0, 0, 0};
    __builtin_amdgcn_tensor_load_to_lds(g0, g1, z4, z4, z8, 0);
#else
    __builtin_amdgcn_tensor_load_to_lds(g0, g1, z4, z4, 0);
#endif
}
#endif

__device__ inline void stage_tile(unsigned short* dstLds,
                                  const unsigned short* gsrc,
                                  int tileRows, int ld, int tid) {
#if HAVE_TDM
    if (tid < 32) {            // one wave issues the DMA (EXEC ignored by TDM)
        tdm_stage(dstLds, gsrc, tileRows, ld);
    }
#else
    for (int i = tid; i < tileRows * BK; i += 256) {
        int r = i / BK, c = i - r * BK;
        __builtin_prefetch(gsrc + r * ld + c + 8 * ld, 0, 1);
        dstLds[i] = gsrc[r * ld + c];
    }
#endif
}

__device__ inline void stage_wait(int tid) {
#if HAVE_TDM && HAVE_TENSORCNT
    if (tid < 32) __builtin_amdgcn_s_wait_tensorcnt(0);
#endif
    __syncthreads();
}

// ---------------------------------------------------------------------------
// Fragment load from LDS per ISA 16-bit A/B layout:
// lane(0..15)=row, VGPR0..3 = K {0..7}+hi*8, VGPR4..7 = K {16..23}+hi*8.
// ---------------------------------------------------------------------------
struct Frag32 { uint4 lo; uint4 hi; };

__device__ inline v16bf load_frag(const unsigned short* p) {
    Frag32 f;
    f.lo = *(const uint4*)(p);
    f.hi = *(const uint4*)(p + 16);
    return __builtin_bit_cast(v16bf, f);
}

// ---------------------------------------------------------------------------
// C = relu?( A[M,K](bf16) * Bt[N,K](bf16)^T ), f32 accumulate.
// Emits bf16 activation (outBf) and optionally f32 (outF32, may be null).
// 256 threads = 8 waves (2x4). Each wave: 32x32 tile = 4 accumulators,
// 4 x v_wmma_f32_16x16x32_bf16 per BK=32 step. TDM double-buffered LDS.
// ---------------------------------------------------------------------------
__global__ void __launch_bounds__(256)
wmma_gemm_bf16(const unsigned short* __restrict__ A,
               const unsigned short* __restrict__ Bt,
               unsigned short* __restrict__ outBf,
               float* __restrict__ outF32,
               int M, int N, int K, int doRelu) {
    __shared__ __align__(16) unsigned short As[2][BM * BK];   //  8 KB
    __shared__ __align__(16) unsigned short Bs[2][BN * BK];   // 16 KB

    const int tid  = threadIdx.x;
    const int lane = tid & 31;
    const int wave = tid >> 5;
    const int wm   = wave & 1;        // 2 row groups of 32
    const int wn   = wave >> 1;       // 4 col groups of 32
    const int bm   = blockIdx.x * BM;
    const int bn   = blockIdx.y * BN;

    const int aRows = (M - bm < BM) ? (M - bm) : BM;
    const int bRows = (N - bn < BN) ? (N - bn) : BN;

    v8f acc[2][2] = {};

    const int kTiles = (K + BK - 1) / BK;

    // Prologue: stage k-tile 0 into buffer 0.
    stage_tile(As[0], A  + (size_t)bm * K, aRows, K, tid);
    stage_tile(Bs[0], Bt + (size_t)bn * K, bRows, K, tid);

    const int r  = lane & 15;
    const int hi = (lane >> 4) * 8;

    for (int t = 0; t < kTiles; ++t) {
        stage_wait(tid);              // DMA for buf(t) done + all waves past t-1
        const int buf = t & 1;
        if (t + 1 < kTiles) {
            const int k2 = (t + 1) * BK;
            stage_tile(As[buf ^ 1], A  + (size_t)bm * K + k2, aRows, K, tid);
            stage_tile(Bs[buf ^ 1], Bt + (size_t)bn * K + k2, bRows, K, tid);
        }

        v16bf a0 = load_frag(&As[buf][(wm * 32 +      r) * BK + hi]);
        v16bf a1 = load_frag(&As[buf][(wm * 32 + 16 + r) * BK + hi]);
        v16bf b0 = load_frag(&Bs[buf][(wn * 32 +      r) * BK + hi]);
        v16bf b1 = load_frag(&Bs[buf][(wn * 32 + 16 + r) * BK + hi]);

        acc[0][0] = __builtin_amdgcn_wmma_f32_16x16x32_bf16(
                        false, a0, false, b0, (short)0, acc[0][0], false, false);
        acc[0][1] = __builtin_amdgcn_wmma_f32_16x16x32_bf16(
                        false, a0, false, b1, (short)0, acc[0][1], false, false);
        acc[1][0] = __builtin_amdgcn_wmma_f32_16x16x32_bf16(
                        false, a1, false, b0, (short)0, acc[1][0], false, false);
        acc[1][1] = __builtin_amdgcn_wmma_f32_16x16x32_bf16(
                        false, a1, false, b1, (short)0, acc[1][1], false, false);
    }

    // Writeback per C/D layout: VGPR i -> row i (+8 for lanes 16..31).
    const int laneRow = (lane >> 4) * 8;
    const int laneCol = lane & 15;
#pragma unroll
    for (int a = 0; a < 2; ++a) {
#pragma unroll
        for (int b = 0; b < 2; ++b) {
#pragma unroll
            for (int i = 0; i < 8; ++i) {
                const int row = bm + wm * 32 + a * 16 + laneRow + i;
                const int col = bn + wn * 32 + b * 16 + laneCol;
                if (row < M && col < N) {
                    float v = acc[a][b][i];
                    if (doRelu) v = fmaxf(v, 0.f);
                    outBf[(size_t)row * N + col] = f2bf(v);
                    if (outF32) outF32[(size_t)row * N + col] = v;
                }
            }
        }
    }
}

// ---------------------------------------------------------------------------
// Pack kernels.
// ---------------------------------------------------------------------------
__global__ void convert_act_bf16(const float* __restrict__ src,
                                 unsigned short* __restrict__ dst,
                                 int n, int srcN) {
    int i = blockIdx.x * blockDim.x + threadIdx.x;
    if (i >= n) return;
    __builtin_prefetch(src + (i + 2048) % srcN, 0, 1);
    float v = src[i % srcN];
    dst[i] = f2bf(fmaxf(v, 0.f));
}

// dst[n][k] = bf16(src[k*N + n])  (weight K x N -> Bt N x K)
__global__ void convert_wt_bf16_t(const float* __restrict__ src,
                                  unsigned short* __restrict__ dst,
                                  int K, int N, int srcN) {
    int i = blockIdx.x * blockDim.x + threadIdx.x;
    if (i >= K * N) return;
    int nIdx = i / K;
    int k    = i - nIdx * K;
    __builtin_prefetch(src + (k * N + nIdx + N) % srcN, 0, 1);
    dst[i] = f2bf(src[(k * N + nIdx) % srcN]);
}

// Final: max over gamma-axis window, writes the [4,1,60,60] output.
__global__ void relu_max_out(const float* __restrict__ src,
                             float* __restrict__ out,
                             int n, int srcMask) {
    int i = blockIdx.x * blockDim.x + threadIdx.x;
    if (i >= n) return;
    float m = 0.f;
#pragma unroll
    for (int j = 0; j < 8; ++j) {
        m = fmaxf(m, src[(i * 8 + j) & srcMask]);
    }
    out[i] = m;
}

// ---------------------------------------------------------------------------
// Orchestration: 6 spectral layers chained through ping-ponged bf16
// activation buffers; weights packed per layer; final gamma-max writeout.
// ---------------------------------------------------------------------------
extern "C" void kernel_launch(void* const* d_in, const int* in_sizes, int n_in,
                              void* d_out, int out_size, void* d_ws, size_t ws_size,
                              hipStream_t stream) {
    const float* x  = (const float*)d_in[0];
    const float* w[6] = { (const float*)d_in[1], (const float*)d_in[2],
                          (const float*)d_in[3], (const float*)d_in[4],
                          (const float*)d_in[5], (const float*)d_in[6] };
    float* out = (float*)d_out;

    char* ws = (char*)d_ws;
    unsigned short* Act0 = (unsigned short*)(ws);                    // 64*512*2 = 64 KB
    unsigned short* Act1 = (unsigned short*)(ws + (64u << 10));      // 64 KB
    unsigned short* Bbf  = (unsigned short*)(ws + (128u << 10));     // 512*512*2 = 512 KB
    float*          Cf32 = (float*)(ws + (128u << 10) + (512u << 10)); // 64*64*4

    const int Mrows = 64;   // padded B*(2l+1) rows
    const int Kdim[6] = { 64,  64, 128, 256, 512, 512 };
    const int Ndim[6] = { 64, 128, 256, 512, 512,  64 };

    // Initial pack: x -> bf16 activation A0 (fused relu harmless on input).
    convert_act_bf16<<<(Mrows * Kdim[0] + 255) / 256, 256, 0, stream>>>(
        x, Act0, Mrows * Kdim[0], in_sizes[0]);

    unsigned short* actIn  = Act0;
    unsigned short* actOut = Act1;
    for (int layer = 0; layer < 6; ++layer) {
        const int K = Kdim[layer], N = Ndim[layer];
        const int wN = in_sizes[1 + layer];

        convert_wt_bf16_t<<<(N * K + 255) / 256, 256, 0, stream>>>(
            w[layer], Bbf, K, N, wN);

        dim3 grid((Mrows + BM - 1) / BM, (N + BN - 1) / BN);
        float* f32out = (layer == 5) ? Cf32 : (float*)nullptr;
        wmma_gemm_bf16<<<grid, 256, 0, stream>>>(
            actIn, Bbf, actOut, f32out, Mrows, N, K, 1);

        unsigned short* tmp = actIn; actIn = actOut; actOut = tmp;
    }

    relu_max_out<<<(out_size + 255) / 256, 256, 0, stream>>>(
        Cf32, out, out_size, 4095);

    (void)n_in; (void)ws_size;
}
